// GGNNClassifierFeatsNoEmb_7842610283354
// MI455X (gfx1250) — matrix operand order repaired and checked by reading, not compiled, not run on hardware
//
#include <hip/hip_runtime.h>
#include <hip/hip_bf16.h>

#define NNODES 100000
#define NEDGES 1600000
#define CH     256
#define NGRAPH 64
#define NBLOCK 5
#define NETYPE 3
#define NSTEP  10

typedef __attribute__((ext_vector_type(16))) __bf16 v16bf;
typedef __attribute__((ext_vector_type(8)))  float  v8f;

__device__ __forceinline__ unsigned short f2bf(float f) {
    unsigned u = __float_as_uint(f);
    unsigned r = (u + 0x7FFFu + ((u >> 16) & 1u)) >> 16;
    return (unsigned short)r;
}
__device__ __forceinline__ unsigned f2ord(float f) {
    unsigned u = __float_as_uint(f);
    return (u & 0x80000000u) ? ~u : (u | 0x80000000u);
}
__device__ __forceinline__ float ord2f(unsigned u) {
    return (u & 0x80000000u) ? __uint_as_float(u & 0x7FFFFFFFu) : __uint_as_float(~u);
}

// ---------------------------------------------------------------- utilities
__global__ void k_zerof(float* p, size_t n) {
    for (size_t i = (size_t)blockIdx.x * blockDim.x + threadIdx.x; i < n;
         i += (size_t)gridDim.x * blockDim.x) p[i] = 0.0f;
}
__global__ void k_copy2(const float* s, const unsigned short* sb,
                        float* d, unsigned short* db, size_t n) {
    for (size_t i = (size_t)blockIdx.x * blockDim.x + threadIdx.x; i < n;
         i += (size_t)gridDim.x * blockDim.x) { d[i] = s[i]; db[i] = sb[i]; }
}
__global__ void k_cvt_bf(const float* s, unsigned short* d, size_t n) {
    for (size_t i = (size_t)blockIdx.x * blockDim.x + threadIdx.x; i < n;
         i += (size_t)gridDim.x * blockDim.x) d[i] = f2bf(s[i]);
}
__global__ void k_add_into(float* d, const float* s, size_t n) {
    for (size_t i = (size_t)blockIdx.x * blockDim.x + threadIdx.x; i < n;
         i += (size_t)gridDim.x * blockDim.x) d[i] += s[i];
}

// ---------------------------------------------------------------- init misc
__global__ void k_init_misc(int* meta, unsigned* gmax, float* denom, float* hg) {
    int i = blockIdx.x * blockDim.x + threadIdx.x;
    if (i < 16)  meta[i] = 0;
    if (i < NGRAPH) { gmax[i] = 0u; denom[i] = 0.0f; }
    if (i < NGRAPH * CH) hg[i] = 0.0f;
}

// ---------------------------------------------------------------- features
__global__ __launch_bounds__(CH)
void k_feat_init(const int* xt, const int* xk, const float* xs,
                 float* h, unsigned short* hb) {
    int i = blockIdx.x, c = threadIdx.x;
    float v;
    if (c < 64) {
        v = (xt[i] == c) ? 1.0f : 0.0f;
    } else if (c < 254) {
        int t = xk[i]; t = t < 0 ? 0 : (t > 189 ? 189 : t);
        v = (t == c - 64) ? 1.0f : 0.0f;
    } else {
        v = xs[(size_t)i * 2 + (c - 254)];
    }
    h[(size_t)i * CH + c]  = v;
    hb[(size_t)i * CH + c] = f2bf(v);
}

// ---------------------------------------------------------------- edge bucketing
__global__ void k_count_edges(const int* et, int* meta, int E) {
    int i = blockIdx.x * blockDim.x + threadIdx.x;
    if (i < E) { int t = et[i]; if (t >= 0 && t < NETYPE) atomicAdd(&meta[t], 1); }
}
__global__ void k_edge_offsets(int* meta) {
    meta[3] = 0; meta[4] = meta[0]; meta[5] = meta[0] + meta[1];
    meta[6] = meta[3]; meta[7] = meta[4]; meta[8] = meta[5];
}
__global__ void k_fill_edges(const int* eidx, const int* et, int* meta,
                             int* esrc, int* edst, int E) {
    int i = blockIdx.x * blockDim.x + threadIdx.x;
    if (i < E) {
        int t = et[i];
        if (t < 0 || t >= NETYPE) return;
        int pos = atomicAdd(&meta[6 + t], 1);
        esrc[pos] = eidx[i];
        edst[pos] = eidx[E + i];
    }
}

// ---------------------------------------------------------------- weight packing
// Pack a K(=256) x Nout f32 matrix (optionally given transposed as Nout x K)
// into this file's wave32 WMMA bf16 B-fragment layout (contiguous k, same as A):
//   fragment (nt,kt): 32 lanes x 8 dwords; lane half selects K 0-15 / 16-31,
//   dword v packs B[kt*32+half*16+2v][n], B[..+1][n]; n = nt*16 + (lane&15).
__global__ void k_pack_b(const float* __restrict__ src, unsigned short* __restrict__ dst,
                         int Nout, int transpose) {
    int j = blockIdx.x * blockDim.x + threadIdx.x;
    int total = (CH * Nout) / 2;
    if (j >= total) return;
    int v    = j & 7;
    int lane = (j >> 3) & 31;
    int frag = j >> 8;
    int kt   = frag & 7;          // K = 256 -> 8 k-tiles
    int nt   = frag >> 3;
    int half = lane >> 4;
    int n    = nt * 16 + (lane & 15);
    int k0   = kt * 32 + half * 16 + 2 * v;
    float f0 = transpose ? src[(size_t)n * CH + k0]       : src[(size_t)k0 * Nout + n];
    float f1 = transpose ? src[(size_t)n * CH + k0 + 1]   : src[(size_t)(k0 + 1) * Nout + n];
    unsigned out = (unsigned)f2bf(f0) | ((unsigned)f2bf(f1) << 16);
    ((unsigned*)dst)[j] = out;
}

// ---------------------------------------------------------------- WMMA GEMM
// C[M x Nout] = A[M x 256](bf16,row-major) @ Bpacked + bias
// block = 256 thr = 8 waves; 32-row A panel in LDS.
// wave -> one 16-row M-tile x FOUR 16-col N-tiles (register blocked):
//   per kt: 2 ds_load_b128 (A, reused 4x) + 8 global_load_b128 (B) + 4 WMMA.
// grid = (M/32, Nout/256)
__global__ __launch_bounds__(256)
void k_wmma_gemm(const unsigned short* __restrict__ A,
                 const unsigned short* __restrict__ Bp,
                 const float* __restrict__ bias,
                 float* __restrict__ Cout, int Nout) {
    __shared__ uint4 lds_a4[1024];                      // 32 rows x 256 bf16 = 16 KB
    unsigned short* lds_a = (unsigned short*)lds_a4;
    const int tid  = threadIdx.x;
    const int row0 = blockIdx.x * 32;
    {
        const uint4* gsrc = (const uint4*)(A + (size_t)row0 * CH);
#pragma unroll
        for (int i = 0; i < 4; ++i) lds_a4[tid + i * 256] = gsrc[tid + i * 256];
    }
    __syncthreads();

    const int wv     = tid >> 5;
    const int lane   = tid & 31;
    const int half   = lane >> 4;
    const int mloc   = lane & 15;
    const int mt     = wv >> 2;                         // 0..1 (M sub-tile)
    const int ntbase = blockIdx.y * 16 + (wv & 3) * 4;  // 4 N-tiles per wave

    v8f acc0 = {}, acc1 = {}, acc2 = {}, acc3 = {};
    // uint4 offset of fragment (nt, kt) for this lane: ((nt*8+kt)*32+lane)*2
    const uint4* b0 = (const uint4*)Bp + (((size_t)ntbase * 8) * 32 + lane) * 2;
#pragma unroll
    for (int kt = 0; kt < 8; ++kt) {
        union { uint4 u[2]; v16bf v; } af, bf0, bf1, bf2, bf3;
        const unsigned short* ap = lds_a + (mt * 16 + mloc) * CH + kt * 32 + half * 16;
        af.u[0] = *(const uint4*)(ap);                  // K = base .. base+7
        af.u[1] = *(const uint4*)(ap + 8);              // K = base+8 .. base+15
        const uint4* bp = b0 + (size_t)kt * 64;         // +kt: 32 lanes * 2 uint4
        bf0.u[0] = bp[0];    bf0.u[1] = bp[1];          // nt+0
        bf1.u[0] = bp[512];  bf1.u[1] = bp[513];        // nt+1 (+8*32*2 uint4)
        bf2.u[0] = bp[1024]; bf2.u[1] = bp[1025];       // nt+2
        bf3.u[0] = bp[1536]; bf3.u[1] = bp[1537];       // nt+3
        __builtin_prefetch(bp + 64, 0, 3);              // next kt fragment (nt+0)
        __builtin_prefetch(bp + 1536 + 64, 0, 3);       // next kt fragment (nt+3)
        acc0 = __builtin_amdgcn_wmma_f32_16x16x32_bf16(
            false, af.v, false, bf0.v, (short)0, acc0, false, false);
        acc1 = __builtin_amdgcn_wmma_f32_16x16x32_bf16(
            false, af.v, false, bf1.v, (short)0, acc1, false, false);
        acc2 = __builtin_amdgcn_wmma_f32_16x16x32_bf16(
            false, af.v, false, bf2.v, (short)0, acc2, false, false);
        acc3 = __builtin_amdgcn_wmma_f32_16x16x32_bf16(
            false, af.v, false, bf3.v, (short)0, acc3, false, false);
    }
    const int rbase = row0 + mt * 16 + half * 8;
#pragma unroll
    for (int j = 0; j < 4; ++j) {
        const int col = (ntbase + j) * 16 + mloc;
        const float bv = bias ? bias[col] : 0.0f;
        const v8f a = (j == 0) ? acc0 : (j == 1) ? acc1 : (j == 2) ? acc2 : acc3;
#pragma unroll
        for (int v = 0; v < 8; ++v)
            Cout[(size_t)(rbase + v) * Nout + col] = a[v] + bv;
    }
}

// ---------------------------------------------------------------- scatter-add
__global__ __launch_bounds__(CH)
void k_scatter_add(const int* esrc, const int* edst, const int* meta, int t,
                   const float* __restrict__ m, float* __restrict__ agg) {
    const int cnt  = meta[t];
    const int base = meta[3 + t];
    const int c    = threadIdx.x;
    for (int e = blockIdx.x; e < cnt; e += gridDim.x) {
        int s = esrc[base + e], d = edst[base + e];
        atomicAdd(&agg[(size_t)d * CH + c], m[(size_t)s * CH + c]);
    }
}

// ---------------------------------------------------------------- GRU elementwise
__global__ __launch_bounds__(CH)
void k_gru_update(const float* __restrict__ gi, const float* __restrict__ gh,
                  float* __restrict__ hc, unsigned short* __restrict__ hcb) {
    int i = blockIdx.x, c = threadIdx.x;
    size_t b3 = (size_t)i * (3 * CH);
    float ir = gi[b3 + c], iz = gi[b3 + CH + c], ig = gi[b3 + 2 * CH + c];
    float hr = gh[b3 + c], hz = gh[b3 + CH + c], hg_ = gh[b3 + 2 * CH + c];
    float r = 1.0f / (1.0f + expf(-(ir + hr)));
    float z = 1.0f / (1.0f + expf(-(iz + hz)));
    float n = tanhf(ig + r * hg_);
    size_t o = (size_t)i * CH + c;
    float hnew = (1.0f - z) * n + z * hc[o];
    hc[o]  = hnew;
    hcb[o] = f2bf(hnew);
}

// ---------------------------------------------------------------- layernorm + leaky
__global__ __launch_bounds__(CH)
void k_ln_leaky(float* __restrict__ h, const float* __restrict__ hmsg,
                const float* __restrict__ g, const float* __restrict__ b,
                unsigned short* __restrict__ hb) {
    __shared__ float s1[CH], s2[CH];
    int i = blockIdx.x, c = threadIdx.x;
    size_t o = (size_t)i * CH + c;
    float x = h[o] + hmsg[o];
    s1[c] = x; s2[c] = x * x;
    __syncthreads();
    for (int s = CH / 2; s > 0; s >>= 1) {
        if (c < s) { s1[c] += s1[c + s]; s2[c] += s2[c + s]; }
        __syncthreads();
    }
    float mu  = s1[0] * (1.0f / CH);
    float var = s2[0] * (1.0f / CH) - mu * mu;
    float y = (x - mu) * rsqrtf(var + 1e-5f) * g[c] + b[c];
    y = y > 0.0f ? y : 0.1f * y;
    h[o]  = y;
    hb[o] = f2bf(y);
}

// ---------------------------------------------------------------- gate MLP (fused)
__global__ __launch_bounds__(128)
void k_gate(const float* __restrict__ h, const float* __restrict__ W1,
            const float* __restrict__ b1, const float* __restrict__ W2,
            const float* __restrict__ b2, float* __restrict__ gate) {
    __shared__ float sh[CH];
    __shared__ float sg[128];
    int i = blockIdx.x, j = threadIdx.x;
    sh[j]       = h[(size_t)i * CH + j];
    sh[j + 128] = h[(size_t)i * CH + j + 128];
    __syncthreads();
    float acc = b1[j];
#pragma unroll 8
    for (int k = 0; k < CH; ++k) acc += sh[k] * W1[k * 128 + j];
    acc = acc > 0.0f ? acc : 0.1f * acc;
    sg[j] = acc * W2[j];
    __syncthreads();
    for (int s = 64; s > 0; s >>= 1) { if (j < s) sg[j] += sg[j + s]; __syncthreads(); }
    if (j == 0) gate[i] = sg[0] + b2[0];
}

// ---------------------------------------------------------------- segment softmax
__global__ void k_seg_max(const float* gate, const int* batch, unsigned* gmax, int N) {
    int i = blockIdx.x * blockDim.x + threadIdx.x;
    if (i < N) atomicMax(&gmax[batch[i]], f2ord(gate[i]));
}
__global__ void k_seg_exp(const float* gate, const int* batch, const unsigned* gmax,
                          float* e, float* denom, int N) {
    int i = blockIdx.x * blockDim.x + threadIdx.x;
    if (i < N) {
        float v = expf(gate[i] - ord2f(gmax[batch[i]]));
        e[i] = v;
        atomicAdd(&denom[batch[i]], v);
    }
}
__global__ __launch_bounds__(CH)
void k_hg_acc(const float* __restrict__ h, const float* __restrict__ e,
              const float* __restrict__ denom, const int* __restrict__ batch,
              float* __restrict__ hg) {
    int i = blockIdx.x, c = threadIdx.x;
    int g = batch[i];
    float a = e[i] / denom[g];
    atomicAdd(&hg[(size_t)g * CH + c], a * h[(size_t)i * CH + c]);
}

// ---------------------------------------------------------------- head MLP
__global__ __launch_bounds__(CH)
void k_head(const float* __restrict__ hg, const float* __restrict__ W1,
            const float* __restrict__ b1, const float* __restrict__ W2,
            const float* __restrict__ b2, float* __restrict__ out) {
    __shared__ float sh[CH];
    __shared__ float so[CH];
    int g = blockIdx.x, j = threadIdx.x;
    sh[j] = hg[(size_t)g * CH + j];
    __syncthreads();
    float acc = b1[j];
#pragma unroll 8
    for (int k = 0; k < CH; ++k) acc += sh[k] * W1[k * CH + j];
    acc = acc > 0.0f ? acc : 0.0f;
    so[j] = acc * W2[j];
    __syncthreads();
    for (int s = CH / 2; s > 0; s >>= 1) { if (j < s) so[j] += so[j + s]; __syncthreads(); }
    if (j == 0) out[g] = so[0] + b2[0];
}

// ================================================================ host
extern "C" void kernel_launch(void* const* d_in, const int* in_sizes, int n_in,
                              void* d_out, int out_size, void* d_ws, size_t ws_size,
                              hipStream_t stream) {
    const int*   x_type   = (const int*)d_in[0];
    const int*   x_tok    = (const int*)d_in[1];
    const float* x_small  = (const float*)d_in[2];
    const int*   eidx     = (const int*)d_in[3];
    const int*   etype    = (const int*)d_in[4];
    const int*   batch    = (const int*)d_in[5];
    const float* conv_W   = (const float*)d_in[6];
    const float* gru_Wih  = (const float*)d_in[7];
    const float* gru_Whh  = (const float*)d_in[8];
    const float* gru_bih  = (const float*)d_in[9];
    const float* gru_bhh  = (const float*)d_in[10];
    const float* ln_g     = (const float*)d_in[11];
    const float* ln_b     = (const float*)d_in[12];
    const float* gate_W1  = (const float*)d_in[13];
    const float* gate_b1  = (const float*)d_in[14];
    const float* gate_W2  = (const float*)d_in[15];
    const float* gate_b2  = (const float*)d_in[16];
    const float* head_W1  = (const float*)d_in[17];
    const float* head_b1  = (const float*)d_in[18];
    const float* head_W2  = (const float*)d_in[19];
    const float* head_b2  = (const float*)d_in[20];
    float* out = (float*)d_out;

    const size_t NC = (size_t)NNODES * CH;

    size_t off = 0;
    auto carve = [&](size_t bytes) -> char* {
        char* p = (char*)d_ws + off;
        off += (bytes + 255) & ~(size_t)255;
        return p;
    };
    float*          h     = (float*)carve(NC * 4);
    float*          hmsg  = (float*)carve(NC * 4);
    float*          hc    = (float*)carve(NC * 4);
    float*          mbuf  = (float*)carve(NC * 4);
    float*          agg   = (float*)carve(NC * 4);
    float*          gi    = (float*)carve(NC * 3 * 4);
    float*          gh    = (float*)carve(NC * 3 * 4);
    unsigned short* hb    = (unsigned short*)carve(NC * 2);
    unsigned short* hcb   = (unsigned short*)carve(NC * 2);
    unsigned short* aggb  = (unsigned short*)carve(NC * 2);
    unsigned short* Wc_p  = (unsigned short*)carve((size_t)NBLOCK * NETYPE * NSTEP * CH * CH * 2);
    unsigned short* Wih_p = (unsigned short*)carve((size_t)NBLOCK * NETYPE * CH * 3 * CH * 2);
    unsigned short* Whh_p = (unsigned short*)carve((size_t)NBLOCK * NETYPE * CH * 3 * CH * 2);
    int*            esrc  = (int*)carve((size_t)NEDGES * 4);
    int*            edst  = (int*)carve((size_t)NEDGES * 4);
    int*            meta  = (int*)carve(16 * 4);
    float*          gatev = (float*)carve((size_t)NNODES * 4);
    float*          ebuf  = (float*)carve((size_t)NNODES * 4);
    unsigned*       gmax  = (unsigned*)carve((size_t)NGRAPH * 4);
    float*          denom = (float*)carve((size_t)NGRAPH * 4);
    float*          hg    = (float*)carve((size_t)NGRAPH * CH * 4);
    (void)ws_size; (void)in_sizes; (void)n_in; (void)out_size;

    const int TB = 256;
    const int gNC = 4096;                       // grid-stride blocks for NC ops
    const int gE  = (NEDGES + TB - 1) / TB;

    // ---- init / features / edge bucketing -------------------------------
    k_init_misc<<<(NGRAPH * CH + TB - 1) / TB, TB, 0, stream>>>(meta, gmax, denom, hg);
    k_feat_init<<<NNODES, CH, 0, stream>>>(x_type, x_tok, x_small, h, hb);
    k_count_edges<<<gE, TB, 0, stream>>>(etype, meta, NEDGES);
    k_edge_offsets<<<1, 1, 0, stream>>>(meta);
    k_fill_edges<<<gE, TB, 0, stream>>>(eidx, etype, meta, esrc, edst, NEDGES);

    // ---- pack weights to bf16 WMMA fragment layout ----------------------
    for (int b = 0; b < NBLOCK; ++b)
        for (int t = 0; t < NETYPE; ++t) {
            int bt = b * NETYPE + t;
            for (int s = 0; s < NSTEP; ++s) {
                size_t o = ((size_t)bt * NSTEP + s) * CH * CH;
                k_pack_b<<<(CH * CH / 2 + TB - 1) / TB, TB, 0, stream>>>(
                    conv_W + o, Wc_p + o, CH, 0);
            }
            size_t wo = (size_t)bt * 3 * CH * CH;
            k_pack_b<<<(CH * 3 * CH / 2 + TB - 1) / TB, TB, 0, stream>>>(
                gru_Wih + wo, Wih_p + wo, 3 * CH, 1);
            k_pack_b<<<(CH * 3 * CH / 2 + TB - 1) / TB, TB, 0, stream>>>(
                gru_Whh + wo, Whh_p + wo, 3 * CH, 1);
        }

    // ---- GGNN blocks ----------------------------------------------------
    const dim3 gemm256(NNODES / 32, 1);   // Nout=256 -> 16 col tiles per block
    const dim3 gemm768(NNODES / 32, 3);   // Nout=768 -> 48 col tiles
    for (int b = 0; b < NBLOCK; ++b) {
        k_zerof<<<gNC, TB, 0, stream>>>(hmsg, NC);
        for (int t = 0; t < NETYPE; ++t) {
            int bt = b * NETYPE + t;
            k_copy2<<<gNC, TB, 0, stream>>>(h, hb, hc, hcb, NC);
            for (int s = 0; s < NSTEP; ++s) {
                // m = hc @ W[b,t,s]
                k_wmma_gemm<<<gemm256, 256, 0, stream>>>(
                    hcb, Wc_p + ((size_t)bt * NSTEP + s) * CH * CH, nullptr, mbuf, CH);
                // agg = scatter_add(m[src] -> dst) over type-t edges
                k_zerof<<<gNC, TB, 0, stream>>>(agg, NC);
                k_scatter_add<<<16384, CH, 0, stream>>>(esrc, edst, meta, t, mbuf, agg);
                k_cvt_bf<<<gNC, TB, 0, stream>>>(agg, aggb, NC);
                // gi = agg @ Wih^T + bih ; gh = hc @ Whh^T + bhh
                k_wmma_gemm<<<gemm768, 256, 0, stream>>>(
                    aggb, Wih_p + (size_t)bt * 3 * CH * CH,
                    gru_bih + (size_t)bt * 3 * CH, gi, 3 * CH);
                k_wmma_gemm<<<gemm768, 256, 0, stream>>>(
                    hcb, Whh_p + (size_t)bt * 3 * CH * CH,
                    gru_bhh + (size_t)bt * 3 * CH, gh, 3 * CH);
                k_gru_update<<<NNODES, CH, 0, stream>>>(gi, gh, hc, hcb);
            }
            k_add_into<<<gNC, TB, 0, stream>>>(hmsg, hc, NC);
        }
        k_ln_leaky<<<NNODES, CH, 0, stream>>>(h, hmsg, ln_g + (size_t)b * CH,
                                              ln_b + (size_t)b * CH, hb);
    }

    // ---- readout --------------------------------------------------------
    k_gate<<<NNODES, 128, 0, stream>>>(h, gate_W1, gate_b1, gate_W2, gate_b2, gatev);
    k_seg_max<<<(NNODES + TB - 1) / TB, TB, 0, stream>>>(gatev, batch, gmax, NNODES);
    k_seg_exp<<<(NNODES + TB - 1) / TB, TB, 0, stream>>>(gatev, batch, gmax, ebuf, denom, NNODES);
    k_hg_acc<<<NNODES, CH, 0, stream>>>(h, ebuf, denom, batch, hg);
    k_head<<<NGRAPH, CH, 0, stream>>>(hg, head_W1, head_b1, head_W2, head_b2, out);
}
